// BinaryNet_21895743275088
// MI455X (gfx1250) — 8-bit matrix operand order repaired and checked by reading, compiled, not run on hardware
//
#include <hip/hip_runtime.h>
#include <cstdint>
#include <cstddef>

#define BN_EPS 1e-3f

typedef __attribute__((ext_vector_type(8))) int v8i;
typedef __attribute__((ext_vector_type(4))) int v4i;
typedef __attribute__((ext_vector_type(2))) int v2i;

__global__ void zero16(int8_t* __restrict__ p, long n16) {
    long i = (long)blockIdx.x * blockDim.x + threadIdx.x;
    if (i >= n16) return;
    v4i z = {0, 0, 0, 0};
    ((v4i*)p)[i] = z;
}

__global__ void prepack_w(const float* __restrict__ w, int8_t* __restrict__ wt,
                          int Ktot, int Cout, int Cpad) {
    long idx = (long)blockIdx.x * blockDim.x + threadIdx.x;
    long total = (long)Cpad * Ktot;
    if (idx >= total) return;
    int co = (int)(idx / Ktot);
    int k  = (int)(idx - (long)co * Ktot);
    int8_t v = 0;
    if (co < Cout) v = (w[(long)k * Cout + co] >= 0.f) ? (int8_t)1 : (int8_t)-1;
    wt[idx] = v;
}

__global__ void conv1_bn_sign(const float* __restrict__ x, const float* __restrict__ w,
                              const float* __restrict__ beta, const float* __restrict__ mean,
                              const float* __restrict__ var, int8_t* __restrict__ out) {
    long idx = (long)blockIdx.x * blockDim.x + threadIdx.x;
    const long total = 256L * 32 * 32 * 128;
    if (idx >= total) return;
    int co = (int)(idx & 127); long t = idx >> 7;
    int xx = (int)(t & 31); t >>= 5;
    int yy = (int)(t & 31); int n = (int)(t >> 5);
    float s = 0.f;
    #pragma unroll
    for (int p = 0; p < 9; ++p) {
        int iy = yy + p / 3 - 1, ix = xx + p % 3 - 1;
        if ((unsigned)iy < 32u && (unsigned)ix < 32u) {
            const float* xp = x + ((long)((n * 32 + iy) * 32 + ix) * 3);
            #pragma unroll
            for (int ci = 0; ci < 3; ++ci) {
                float wv = (w[(p * 3 + ci) * 128 + co] >= 0.f) ? 1.f : -1.f;
                s += xp[ci] * wv;
            }
        }
    }
    float val = (s - mean[co]) * rsqrtf(var[co] + BN_EPS) + beta[co];
    size_t oidx = (((size_t)(n * 34 + yy + 1)) * 34 + xx + 1) * 128 + co;
    out[oidx] = (val >= 0.f) ? (int8_t)1 : (int8_t)-1;
}

static __device__ __forceinline__ v8i load_a_frag(const int8_t* base, int g) {
    v8i a;
    #pragma unroll
    for (int p = 0; p < 2; ++p) {
        v4i t = *(const v4i*)(base + (p << 5) + (g << 4));
        a[4 * p]     = t.x;
        a[4 * p + 1] = t.y;
        a[4 * p + 2] = t.z;
        a[4 * p + 3] = t.w;
    }
    return a;
}

static __device__ __forceinline__ v8i load_b_frag(const int8_t* base, int g) {
    v4i t0 = *(const v4i*)(base + (g << 4));
    v4i t1 = *(const v4i*)(base + 32 + (g << 4));
    v8i b;
    b[0] = t0.x; b[1] = t0.y; b[2] = t0.z; b[3] = t0.w;
    b[4] = t1.x; b[5] = t1.y; b[6] = t1.z; b[7] = t1.w;
    return b;
}

__global__ void __launch_bounds__(256) bgemm_iu8(
    const int8_t* __restrict__ act, const int8_t* __restrict__ wt,
    short* __restrict__ out, int H, int W, int Cin, int Ktot, int Cpad,
    int Mtot, int conv3x3) {

    int wave   = blockIdx.x * (blockDim.x >> 5) + (threadIdx.x >> 5);
    int ntiles = Cpad >> 6;
    int mtiles = Mtot >> 5;
    if (wave >= mtiles * ntiles) return;
    int mt = wave / ntiles, ntb = wave - mt * ntiles;
    int m0 = mt << 5, n0 = ntb << 6;

    int lane = threadIdx.x & 31;
    int g  = lane >> 4;
    int hm = lane & 15;

    int pm0 = m0 + hm;
    int pm1 = m0 + 16 + hm;

    int ni0 = 0, y0 = 0, x0 = 0, ni1 = 0, y1 = 0, x1 = 0;
    if (conv3x3) {
        int hw = H * W;
        ni0 = pm0 / hw; int r0 = pm0 - ni0 * hw; y0 = r0 / W; x0 = r0 - y0 * W;
        ni1 = pm1 / hw; int r1 = pm1 - ni1 * hw; y1 = r1 / W; x1 = r1 - y1 * W;
    }
    int Hp = H + 2, Wp = W + 2;

    int P    = conv3x3 ? 9 : 1;
    int Klen = conv3x3 ? Cin : Ktot;
    const int8_t* wrow = wt + (size_t)(n0 + hm) * Ktot;

    v8i acc[8] = {};

    for (int p = 0; p < P; ++p) {
        const int8_t *ab0, *ab1;
        if (conv3x3) {
            int ky = p / 3, kx = p - ky * 3;
            ab0 = act + (size_t)((ni0 * Hp + y0 + ky) * Wp + (x0 + kx)) * Cin;
            ab1 = act + (size_t)((ni1 * Hp + y1 + ky) * Wp + (x1 + kx)) * Cin;
        } else {
            ab0 = act + (size_t)pm0 * Ktot;
            ab1 = act + (size_t)pm1 * Ktot;
        }
        const int8_t* bp = wrow + (size_t)p * Klen;

        for (int c0 = 0; c0 < Klen; c0 += 64) {
            v8i a0 = load_a_frag(ab0 + c0, g);
            v8i a1 = load_a_frag(ab1 + c0, g);
            #pragma unroll
            for (int t = 0; t < 4; ++t) {
                v8i b = load_b_frag(bp + c0 + (size_t)(t << 4) * Ktot, g);
                acc[t]     = __builtin_amdgcn_wmma_i32_16x16x64_iu8(true, a0, true, b, acc[t],     false, false);
                acc[t + 4] = __builtin_amdgcn_wmma_i32_16x16x64_iu8(true, a1, true, b, acc[t + 4], false, false);
            }
        }
    }

    #pragma unroll
    for (int d = 0; d < 8; ++d) {
        size_t r0 = (size_t)(m0 + d + (g << 3)) * Cpad + n0 + hm;
        size_t r1 = r0 + (size_t)16 * Cpad;
        out[r0]      = (short)acc[0][d];
        out[r0 + 16] = (short)acc[1][d];
        out[r0 + 32] = (short)acc[2][d];
        out[r0 + 48] = (short)acc[3][d];
        out[r1]      = (short)acc[4][d];
        out[r1 + 16] = (short)acc[5][d];
        out[r1 + 32] = (short)acc[6][d];
        out[r1 + 48] = (short)acc[7][d];
    }
}

__global__ void bn_sign(const short* __restrict__ cbuf, const float* __restrict__ beta,
                        const float* __restrict__ mean, const float* __restrict__ var,
                        int8_t* __restrict__ out, int N, int H, int W, int C, int pad) {
    long total = (long)N * H * W * C;
    long idx = (long)blockIdx.x * blockDim.x + threadIdx.x;
    if (idx >= total) return;
    int c = (int)(idx & (C - 1)); long t = idx / C;
    int xp = (int)(t % W); t /= W;
    int yp = (int)(t % H); int n = (int)(t / H);
    float v = ((float)cbuf[idx] - mean[c]) * rsqrtf(var[c] + BN_EPS) + beta[c];
    int Hp = H + 2 * pad, Wp = W + 2 * pad;
    size_t oidx = (((size_t)n * Hp + yp + pad) * Wp + xp + pad) * (size_t)C + c;
    out[oidx] = (v >= 0.f) ? (int8_t)1 : (int8_t)-1;
}

__global__ void pool_bn_sign(const short* __restrict__ cbuf, const float* __restrict__ beta,
                             const float* __restrict__ mean, const float* __restrict__ var,
                             int8_t* __restrict__ out, int N, int H, int W, int C, int pad) {
    int H2 = H >> 1, W2 = W >> 1;
    long total = (long)N * H2 * W2 * C;
    long idx = (long)blockIdx.x * blockDim.x + threadIdx.x;
    if (idx >= total) return;
    int c = (int)(idx & (C - 1)); long t = idx / C;
    int xo = (int)(t % W2); t /= W2;
    int yo = (int)(t % H2); int n = (int)(t / H2);
    size_t base = ((size_t)(n * H + 2 * yo) * W + 2 * xo) * C + c;
    size_t rowS = (size_t)W * C;
    int m01 = max((int)cbuf[base],        (int)cbuf[base + C]);
    int m23 = max((int)cbuf[base + rowS], (int)cbuf[base + rowS + C]);
    int mm  = max(m01, m23);
    float v = ((float)mm - mean[c]) * rsqrtf(var[c] + BN_EPS) + beta[c];
    int Hp = H2 + 2 * pad, Wp = W2 + 2 * pad;
    size_t oidx = (((size_t)n * Hp + yo + pad) * Wp + xo + pad) * (size_t)C + c;
    out[oidx] = (v >= 0.f) ? (int8_t)1 : (int8_t)-1;
}

__global__ void bn_softmax10(const short* __restrict__ cbuf, const float* __restrict__ beta,
                             const float* __restrict__ mean, const float* __restrict__ var,
                             float* __restrict__ out) {
    int row = blockIdx.x * blockDim.x + threadIdx.x;
    if (row >= 256) return;
    float l[10]; float mx = -1e30f;
    #pragma unroll
    for (int c = 0; c < 10; ++c) {
        float v = ((float)cbuf[row * 64 + c] - mean[c]) * rsqrtf(var[c] + BN_EPS) + beta[c];
        l[c] = v; mx = fmaxf(mx, v);
    }
    float s = 0.f;
    #pragma unroll
    for (int c = 0; c < 10; ++c) { l[c] = expf(l[c] - mx); s += l[c]; }
    float inv = 1.f / s;
    #pragma unroll
    for (int c = 0; c < 10; ++c) out[row * 10 + c] = l[c] * inv;
}

static inline unsigned gblk(long n) { return (unsigned)((n + 255) / 256); }

extern "C" void kernel_launch(void* const* d_in, const int* in_sizes, int n_in,
                              void* d_out, int out_size, void* d_ws, size_t ws_size,
                              hipStream_t stream) {
    const float* x    = (const float*)d_in[0];
    const float* w11  = (const float*)d_in[1];
    const float *b11 = (const float*)d_in[2],  *m11 = (const float*)d_in[3],  *v11 = (const float*)d_in[4];
    const float* w12  = (const float*)d_in[5];
    const float *b12 = (const float*)d_in[6],  *m12 = (const float*)d_in[7],  *v12 = (const float*)d_in[8];
    const float* w21  = (const float*)d_in[9];
    const float *b21 = (const float*)d_in[10], *m21 = (const float*)d_in[11], *v21 = (const float*)d_in[12];
    const float* w22  = (const float*)d_in[13];
    const float *b22 = (const float*)d_in[14], *m22 = (const float*)d_in[15], *v22 = (const float*)d_in[16];
    const float* w31  = (const float*)d_in[17];
    const float *b31 = (const float*)d_in[18], *m31 = (const float*)d_in[19], *v31 = (const float*)d_in[20];
    const float* w32  = (const float*)d_in[21];
    const float *b32 = (const float*)d_in[22], *m32 = (const float*)d_in[23], *v32 = (const float*)d_in[24];
    const float* wf5  = (const float*)d_in[25];
    const float *b5  = (const float*)d_in[26], *m5  = (const float*)d_in[27], *v5  = (const float*)d_in[28];
    const float* wf6  = (const float*)d_in[29];
    const float *b6  = (const float*)d_in[30], *m6  = (const float*)d_in[31], *v6  = (const float*)d_in[32];
    const float* wf7  = (const float*)d_in[33];
    const float *b7  = (const float*)d_in[34], *m7  = (const float*)d_in[35], *v7  = (const float*)d_in[36];

    char* ws = (char*)d_ws;
    size_t o = 0;
    auto take = [&](size_t bytes) { size_t r = o; o += (bytes + 255) & ~(size_t)255; return r; };
    int8_t* wt12 = (int8_t*)(ws + take((size_t)128 * 1152));
    int8_t* wt21 = (int8_t*)(ws + take((size_t)256 * 1152));
    int8_t* wt22 = (int8_t*)(ws + take((size_t)256 * 2304));
    int8_t* wt31 = (int8_t*)(ws + take((size_t)512 * 2304));
    int8_t* wt32 = (int8_t*)(ws + take((size_t)512 * 4608));
    int8_t* wtf5 = (int8_t*)(ws + take((size_t)1024 * 8192));
    int8_t* wtf6 = (int8_t*)(ws + take((size_t)1024 * 1024));
    int8_t* wtf7 = (int8_t*)(ws + take((size_t)64 * 1024));
    const size_t szA = (size_t)256 * 34 * 34 * 128;
    const size_t szB = (size_t)256 * 18 * 18 * 128;
    int8_t* actA = (int8_t*)(ws + take(szA));
    int8_t* actB = (int8_t*)(ws + take(szB));
    short*  cbuf = (short*) (ws + take((size_t)256 * 32 * 32 * 128 * 2));

    prepack_w<<<gblk((long)128 * 1152),  256, 0, stream>>>(w12, wt12, 1152, 128, 128);
    prepack_w<<<gblk((long)256 * 1152),  256, 0, stream>>>(w21, wt21, 1152, 256, 256);
    prepack_w<<<gblk((long)256 * 2304),  256, 0, stream>>>(w22, wt22, 2304, 256, 256);
    prepack_w<<<gblk((long)512 * 2304),  256, 0, stream>>>(w31, wt31, 2304, 512, 512);
    prepack_w<<<gblk((long)512 * 4608),  256, 0, stream>>>(w32, wt32, 4608, 512, 512);
    prepack_w<<<gblk((long)1024 * 8192), 256, 0, stream>>>(wf5, wtf5, 8192, 1024, 1024);
    prepack_w<<<gblk((long)1024 * 1024), 256, 0, stream>>>(wf6, wtf6, 1024, 1024, 1024);
    prepack_w<<<gblk((long)64 * 1024),   256, 0, stream>>>(wf7, wtf7, 1024, 10, 64);

    auto gemm = [&](const int8_t* act, const int8_t* wt, int H, int W, int Cin,
                    int Ktot, int Cpad, int Mtot, int conv) {
        long waves = (long)(Mtot >> 5) * (Cpad >> 6);
        bgemm_iu8<<<(unsigned)((waves + 7) / 8), 256, 0, stream>>>(
            act, wt, cbuf, H, W, Cin, Ktot, Cpad, Mtot, conv);
    };
    auto zero = [&](int8_t* p, size_t bytes) {
        zero16<<<gblk((long)(bytes >> 4)), 256, 0, stream>>>(p, (long)(bytes >> 4));
    };

    zero(actA, (size_t)256 * 34 * 34 * 128);
    conv1_bn_sign<<<gblk(256L * 32 * 32 * 128), 256, 0, stream>>>(x, w11, b11, m11, v11, actA);
    gemm(actA, wt12, 32, 32, 128, 1152, 128, 256 * 32 * 32, 1);
    zero(actB, (size_t)256 * 18 * 18 * 128);
    pool_bn_sign<<<gblk(256L * 16 * 16 * 128), 256, 0, stream>>>(cbuf, b12, m12, v12, actB, 256, 32, 32, 128, 1);
    gemm(actB, wt21, 16, 16, 128, 1152, 256, 256 * 16 * 16, 1);
    zero(actA, (size_t)256 * 18 * 18 * 256);
    bn_sign<<<gblk(256L * 16 * 16 * 256), 256, 0, stream>>>(cbuf, b21, m21, v21, actA, 256, 16, 16, 256, 1);
    gemm(actA, wt22, 16, 16, 256, 2304, 256, 256 * 16 * 16, 1);
    zero(actB, (size_t)256 * 10 * 10 * 256);
    pool_bn_sign<<<gblk(256L * 8 * 8 * 256), 256, 0, stream>>>(cbuf, b22, m22, v22, actB, 256, 16, 16, 256, 1);
    gemm(actB, wt31, 8, 8, 256, 2304, 512, 256 * 8 * 8, 1);
    zero(actA, (size_t)256 * 10 * 10 * 512);
    bn_sign<<<gblk(256L * 8 * 8 * 512), 256, 0, stream>>>(cbuf, b31, m31, v31, actA, 256, 8, 8, 512, 1);
    gemm(actA, wt32, 8, 8, 512, 4608, 512, 256 * 8 * 8, 1);
    pool_bn_sign<<<gblk(256L * 4 * 4 * 512), 256, 0, stream>>>(cbuf, b32, m32, v32, actB, 256, 8, 8, 512, 0);
    gemm(actB, wtf5, 1, 1, 8192, 8192, 1024, 256, 0);
    bn_sign<<<gblk(256L * 1024), 256, 0, stream>>>(cbuf, b5, m5, v5, actA, 256, 1, 1, 1024, 0);
    gemm(actA, wtf6, 1, 1, 1024, 1024, 1024, 256, 0);
    bn_sign<<<gblk(256L * 1024), 256, 0, stream>>>(cbuf, b6, m6, v6, actB, 256, 1, 1, 1024, 0);
    gemm(actB, wtf7, 1, 1, 1024, 1024, 64, 256, 0);
    bn_softmax10<<<1, 256, 0, stream>>>(cbuf, b7, m7, v7, (float*)d_out);
}
